// CrossAttentionValueFuser_7138235646348
// MI455X (gfx1250) — matrix operand order repaired and checked by reading, not compiled, not run on hardware
//
#include <hip/hip_runtime.h>

typedef __attribute__((ext_vector_type(16))) _Float16 v16h;
typedef __attribute__((ext_vector_type(8)))  float    v8f;
typedef __attribute__((ext_vector_type(4)))  unsigned int v4u;
typedef __attribute__((ext_vector_type(4)))  int      v4i;
typedef __attribute__((ext_vector_type(8)))  int      v8i;

#define NB    4
#define CIN   512     // Cx == Cf
#define CH    256     // hidden channels
#define CF    512     // value channels
#define PPIX  4096    // H*W

union Frag {
  v16h  h;
  uint4 q[2];
  _Float16 e[16];
};

__device__ __forceinline__ v8f wmma_f16(const Frag& a, const Frag& b, v8f c) {
  return __builtin_amdgcn_wmma_f32_16x16x32_f16(false, a.h, false, b.h,
                                                (short)0, c, false, false);
}

__device__ __forceinline__ float redmax16(float v) {
  v = fmaxf(v, __shfl_xor(v, 1, 32));
  v = fmaxf(v, __shfl_xor(v, 2, 32));
  v = fmaxf(v, __shfl_xor(v, 4, 32));
  v = fmaxf(v, __shfl_xor(v, 8, 32));
  return v;
}
__device__ __forceinline__ float redsum16(float v) {
  v += __shfl_xor(v, 1, 32);
  v += __shfl_xor(v, 2, 32);
  v += __shfl_xor(v, 4, 32);
  v += __shfl_xor(v, 8, 32);
  return v;
}

// ---- CDNA5 async global->LDS copy (ASYNCcnt-tracked), 16B per lane --------
__device__ __forceinline__ void async_copy_b128(unsigned int lds_addr,
                                                const void* gaddr) {
  asm volatile("global_load_async_to_lds_b128 %0, %1, off"
               :: "v"(lds_addr), "v"(gaddr) : "memory");
}
// wait until at most N async ops remain outstanding (in-order completion)
__device__ __forceinline__ void wait_asynccnt8() {
  asm volatile("s_wait_asynccnt 8" ::: "memory");
}
__device__ __forceinline__ void wait_asynccnt0() {
  asm volatile("s_wait_asynccnt 0" ::: "memory");
}
__device__ __forceinline__ unsigned int lds_off(const void* p) {
  return (unsigned int)(unsigned long long)p;   // low 32 bits = LDS byte addr
}

// ---- CDNA5 Tensor Data Mover: DMA a 2D f16 tile (row-padded) into LDS -----
// LDS gets +8 DWORDs padding per 32 DWORDs (=> 80-half row stride).
__device__ __forceinline__ void tdm_load_2d_f16(unsigned int lds_addr,
                                                const void* gaddr,
                                                unsigned int tsr_d0,
                                                unsigned int tsr_d1,
                                                unsigned int tile_d0,
                                                unsigned int tile_d1,
                                                unsigned int stride_d0) {
  unsigned long long ga = (unsigned long long)gaddr;
  v4u g0;
  g0[0] = 1u;                                            // count=1, user desc
  g0[1] = lds_addr;                                      // lds_addr (bytes)
  g0[2] = (unsigned int)ga;                              // global_addr[31:0]
  g0[3] = ((unsigned int)(ga >> 32) & 0x01FFFFFFu) | (2u << 30); // [56:32]|type=2
  v8i g1;
  g1[0] = (int)((1u << 16)      // data_size = 2 bytes
              | (1u << 20)      // pad_enable
              | (4u << 22)      // pad_interval: 32 DWORDs (64 halves)
              | (7u << 25));    // pad_amount:   8 DWORDs (16 halves)
  g1[1] = (int)((tsr_d0 & 0xFFFFu) << 16);               // tensor_dim0[15:0]
  g1[2] = (int)(((tsr_d0 >> 16) & 0xFFFFu) | ((tsr_d1 & 0xFFFFu) << 16));
  g1[3] = (int)(((tsr_d1 >> 16) & 0xFFFFu) | ((tile_d0 & 0xFFFFu) << 16));
  g1[4] = (int)(tile_d1 & 0xFFFFu);                      // tile_dim1, tile_dim2=0
  g1[5] = (int)stride_d0;                                // tensor_dim0_stride[31:0]
  g1[6] = 0;
  g1[7] = 0;
  v4i z4 = {0, 0, 0, 0};
#if defined(__clang_major__) && (__clang_major__ >= 23)
  v8i z8 = {0, 0, 0, 0, 0, 0, 0, 0};
  __builtin_amdgcn_tensor_load_to_lds(g0, g1, z4, z4, z8, 0);
#else
  __builtin_amdgcn_tensor_load_to_lds(g0, g1, z4, z4, 0);
#endif
}

// ---------------------------------------------------------------------------
// Stage 1: Out[b][p][co] = (f16)( sum_ci X[b][ci][p] * Wh[co][ci] + bias[co] )
// Wh is pre-converted f16, so B-fragments are two contiguous b128 loads.
// ---------------------------------------------------------------------------
__global__ __launch_bounds__(256) void proj_kernel(const float* __restrict__ X,
                                                   const _Float16* __restrict__ Wh,
                                                   const float* __restrict__ bias,
                                                   _Float16* __restrict__ Out) {
  __shared__ _Float16 Xt[128 * 528];          // [p][ci], padded row stride
  const int b     = blockIdx.y;
  const int pBase = blockIdx.x * 128;
  const int tid   = threadIdx.x;

  // stage + transpose X tile (coalesced f32 reads along p)
  {
    const int p   = tid & 127;
    const int ci0 = tid >> 7;                 // 0..1
    for (int ci = ci0; ci < CIN; ci += 2) {
      float v = X[((size_t)(b * CIN + ci)) * PPIX + pBase + p];
      Xt[p * 528 + ci] = (_Float16)v;
    }
  }
  __syncthreads();

  const int lane = tid & 31;
  const int w    = tid >> 5;                  // wave = 16-row strip
  const int hs   = lane >> 4;
  const int ln   = lane & 15;

  // hold all 16 A-fragments (K = 512) in VGPRs
  Frag A[16];
  {
    const _Float16* rowp = &Xt[(w * 16 + ln) * 528];
    #pragma unroll
    for (int kc = 0; kc < 16; ++kc) {
      const _Float16* bp = rowp + kc * 32 + hs * 8;
      A[kc].q[0] = *(const uint4*)(bp);
      A[kc].q[1] = *(const uint4*)(bp + 16);
    }
  }

  for (int nt = 0; nt < 16; ++nt) {
    const int co = nt * 16 + ln;
    v8f acc = {};
    #pragma unroll
    for (int kc = 0; kc < 16; ++kc) {
      Frag Bf;                                // B[k][n] = Wh[co=n][ci=k]
      const _Float16* bp = Wh + (size_t)co * CIN + kc * 32 + hs * 8;
      Bf.q[0] = *(const uint4*)bp;
      Bf.q[1] = *(const uint4*)(bp + 16);
      acc = wmma_f16(A[kc], Bf, acc);
    }
    const float bv = bias[co];
    #pragma unroll
    for (int r = 0; r < 8; ++r) {
      const int p = pBase + w * 16 + hs * 8 + r;
      Out[((size_t)b * PPIX + p) * CH + co] = (_Float16)(acc[r] + bv);
    }
  }
}

// ---------------------------------------------------------------------------
// Stage 2: f32 -> f16 bulk convert (V tensor and weight matrices)
// ---------------------------------------------------------------------------
__global__ __launch_bounds__(256) void half_cvt(const float* __restrict__ in,
                                                _Float16* __restrict__ out,
                                                int n4) {
  int i = blockIdx.x * 256 + threadIdx.x;
  if (i < n4) {
    float4 v = ((const float4*)in)[i];
    union { _Float16 h[4]; uint2 u; } pk;
    pk.h[0] = (_Float16)v.x; pk.h[1] = (_Float16)v.y;
    pk.h[2] = (_Float16)v.z; pk.h[3] = (_Float16)v.w;
    ((uint2*)out)[i] = pk.u;
  }
}

// ---------------------------------------------------------------------------
// Stage 3: flash attention, double-buffered K/V tiles.
// Tile i+1 is issued (async b128 for K, one TDM op for V) before waiting on
// tile i: in-order ASYNCcnt/TENSORcnt lets "wait <= 8" / "wait <= 1" retire
// exactly the older tile while the next one streams in behind compute.
// ---------------------------------------------------------------------------
__global__ __launch_bounds__(256) void flash_kernel(const _Float16* __restrict__ Qh,
                                                    const _Float16* __restrict__ Kh,
                                                    const _Float16* __restrict__ Vh,
                                                    float* __restrict__ Out) {
  __shared__ _Float16 Qt[64 * 272];           // [q][ch]
  __shared__ _Float16 Kt[2][64 * 272];        // [key][ch], double buffered
  __shared__ _Float16 Vt[2][512 * 80];        // [c][key], 80 = 64 + TDM pad
  __shared__ _Float16 Pt[64 * 80];            // [q][key]
  __shared__ float partMax[2][64];
  __shared__ float partSum[2][64];
  __shared__ float rowM[64];
  __shared__ float rowL[64];
  __shared__ float rowAlpha[64];

  const int b     = blockIdx.y;
  const int qBase = blockIdx.x * 64;
  const int tid   = threadIdx.x;
  const int lane  = tid & 31;
  const int w     = tid >> 5;
  const int hs    = lane >> 4;
  const int ln    = lane & 15;
  const int strip   = w & 3;                  // 16-row strip for S stage
  const int colHalf = w >> 2;                 // 32-key half for S stage

  const _Float16* Ksrc = Kh + (size_t)b * PPIX * CH;
  const _Float16* Vsrc = Vh + (size_t)b * CF * PPIX;

  // prologue: Q tile async (8/lane), then tile 0 of K (async) and V (TDM)
  {
    const _Float16* src = Qh + ((size_t)b * PPIX + qBase) * CH;
    for (int i = tid; i < 64 * 32; i += 256) {
      int r = i >> 5, jc = i & 31;
      async_copy_b128(lds_off(&Qt[r * 272 + jc * 8]), src + i * 8);
    }
  }
  for (int i = tid; i < 64 * 32; i += 256) {
    int r = i >> 5, jc = i & 31;
    async_copy_b128(lds_off(&Kt[0][r * 272 + jc * 8]), Ksrc + i * 8);
  }
  if (w == 0)
    tdm_load_2d_f16(lds_off(&Vt[0][0]), Vsrc, PPIX, CF, 64, CF, PPIX);
  if (tid < 64) { rowM[tid] = -__builtin_inff(); rowL[tid] = 0.f; }
  wait_asynccnt8();                           // Q landed (K0 may be in flight)
  __syncthreads();

  // Q A-fragments for this wave's strip, kept in VGPRs for all 64 steps
  Frag QA[8];
  {
    const _Float16* rowp = &Qt[(strip * 16 + ln) * 272];
    #pragma unroll
    for (int kc = 0; kc < 8; ++kc) {
      const _Float16* bp = rowp + kc * 32 + hs * 8;
      QA[kc].q[0] = *(const uint4*)(bp);
      QA[kc].q[1] = *(const uint4*)(bp + 16);
    }
  }

  v8f O[4][4] = {};                           // 64 rows x 64 channels, fp32

  for (int it = 0; it < PPIX / 64; ++it) {
    const int buf = it & 1;
    __syncthreads();                          // B0: prior step done with buf^1

    if (it + 1 < PPIX / 64) {                 // issue tile it+1 into buf^1
      const int kvn = (it + 1) * 64;
      for (int i = tid; i < 64 * 32; i += 256) {
        int r = i >> 5, jc = i & 31;
        async_copy_b128(lds_off(&Kt[buf ^ 1][r * 272 + jc * 8]),
                        Ksrc + (size_t)kvn * CH + i * 8);
      }
      if (w == 0)
        tdm_load_2d_f16(lds_off(&Vt[buf ^ 1][0]), Vsrc + kvn,
                        PPIX, CF, 64, CF, PPIX);
      if (it + 2 < PPIX / 64)                 // warm L2 for the tile after
        __builtin_prefetch(Ksrc + (size_t)(it + 2) * 64 * CH + tid * 64, 0, 0);
      wait_asynccnt8();                       // tile it retired, it+1 in flight
      if (w == 0) __builtin_amdgcn_s_wait_tensorcnt(1);
    } else {
      wait_asynccnt0();
      if (w == 0) __builtin_amdgcn_s_wait_tensorcnt(0);
    }
    __syncthreads();                          // B1: tile it visible everywhere

    // ---- S = Q K^T : wave covers [strip*16 .. +16) x [colHalf*32 .. +32)
    v8f S[2];
    #pragma unroll
    for (int ct = 0; ct < 2; ++ct) {
      v8f acc = {};
      #pragma unroll
      for (int kc = 0; kc < 8; ++kc) {
        Frag Bf;                              // B[k][n] = K[key=n][ch=k]
        const _Float16* bp =
            &Kt[buf][(colHalf * 32 + ct * 16 + ln) * 272 + kc * 32 + hs * 8];
        Bf.q[0] = *(const uint4*)bp;
        Bf.q[1] = *(const uint4*)(bp + 16);
        acc = wmma_f16(QA[kc], Bf, acc);
      }
      S[ct] = acc;
    }

    // per-row max of this wave's 32 columns
    #pragma unroll
    for (int r = 0; r < 8; ++r) {
      float mx = redmax16(fmaxf(S[0][r], S[1][r]));
      if (ln == 0) partMax[colHalf][strip * 16 + hs * 8 + r] = mx;
    }
    __syncthreads();                          // B2

    if (w == 0) {                             // online max + alpha
      for (int rr = lane; rr < 64; rr += 32) {
        float mo = rowM[rr];
        float mn = fmaxf(mo, fmaxf(partMax[0][rr], partMax[1][rr]));
        rowM[rr] = mn;
        rowAlpha[rr] = __expf(mo - mn);
      }
    }
    __syncthreads();                          // B3

    // P = exp(S - m), row partial sums, stage P as f16
    #pragma unroll
    for (int r = 0; r < 8; ++r) {
      const int row = strip * 16 + hs * 8 + r;
      const float mrow = rowM[row];
      float p0 = __expf(S[0][r] - mrow);
      float p1 = __expf(S[1][r] - mrow);
      Pt[row * 80 + colHalf * 32 + ln]      = (_Float16)p0;
      Pt[row * 80 + colHalf * 32 + 16 + ln] = (_Float16)p1;
      float sm = redsum16(p0 + p1);
      if (ln == 0) partSum[colHalf][row] = sm;
    }
    __syncthreads();                          // B4

    if (w == 0) {                             // online denominator
      for (int rr = lane; rr < 64; rr += 32)
        rowL[rr] = rowL[rr] * rowAlpha[rr] + partSum[0][rr] + partSum[1][rr];
    }

    // ---- O = O*alpha + P V : wave owns channels [w*64 .. +64)
    #pragma unroll
    for (int s = 0; s < 4; ++s) {
      float a8[8];
      #pragma unroll
      for (int r = 0; r < 8; ++r) a8[r] = rowAlpha[s * 16 + hs * 8 + r];
      #pragma unroll
      for (int c = 0; c < 4; ++c)
        #pragma unroll
        for (int r = 0; r < 8; ++r) O[s][c][r] *= a8[r];
    }
    const int cBase = w * 64;
    #pragma unroll
    for (int kc = 0; kc < 2; ++kc) {
      Frag PA[4];
      #pragma unroll
      for (int s = 0; s < 4; ++s) {
        const _Float16* bp = &Pt[(s * 16 + ln) * 80 + kc * 32 + hs * 8];
        PA[s].q[0] = *(const uint4*)bp;
        PA[s].q[1] = *(const uint4*)(bp + 16);
      }
      #pragma unroll
      for (int c = 0; c < 4; ++c) {
        Frag Bf;                              // B[k][n] = V[c=n][key=k]
        const _Float16* bp =
            &Vt[buf][(cBase + c * 16 + ln) * 80 + kc * 32 + hs * 8];
        Bf.q[0] = *(const uint4*)bp;
        Bf.q[1] = *(const uint4*)(bp + 16);
        #pragma unroll
        for (int s = 0; s < 4; ++s) O[s][c] = wmma_f16(PA[s], Bf, O[s][c]);
      }
    }
  }
  __syncthreads();

  // epilogue: O / l  ->  Out[b][c][p]
  #pragma unroll
  for (int s = 0; s < 4; ++s) {
    float linv[8];
    #pragma unroll
    for (int r = 0; r < 8; ++r) linv[r] = 1.0f / rowL[s * 16 + hs * 8 + r];
    #pragma unroll
    for (int c = 0; c < 4; ++c) {
      const int cg = w * 64 + c * 16 + ln;
      float* dst = Out + ((size_t)b * CF + cg) * PPIX + qBase + s * 16 + hs * 8;
      #pragma unroll
      for (int r = 0; r < 8; ++r) dst[r] = O[s][c][r] * linv[r];
    }
  }
}

// ---------------------------------------------------------------------------
extern "C" void kernel_launch(void* const* d_in, const int* in_sizes, int n_in,
                              void* d_out, int out_size, void* d_ws, size_t ws_size,
                              hipStream_t stream) {
  const float* X  = (const float*)d_in[0];   // query_features  [4,512,64,64]
  const float* F  = (const float*)d_in[1];   // reference_features
  const float* Wq = (const float*)d_in[2];   // [256,512]
  const float* bq = (const float*)d_in[3];   // [256]
  const float* Wk = (const float*)d_in[4];
  const float* bk = (const float*)d_in[5];
  float* out = (float*)d_out;                // [4,512,64,64]

  _Float16* Qh  = (_Float16*)d_ws;                      // [b][p][ch]  8 MB
  _Float16* Kh  = Qh  + (size_t)NB * PPIX * CH;         // [b][p][ch]  8 MB
  _Float16* Vh  = Kh  + (size_t)NB * PPIX * CH;         // [b][c][p]  16 MB
  _Float16* Wqh = Vh  + (size_t)NB * CF * PPIX;         // [256][512] 256 KB
  _Float16* Wkh = Wqh + (size_t)CH * CIN;               // [256][512] 256 KB

  const int nW4 = CH * CIN / 4;
  half_cvt<<<(nW4 + 255) / 256, 256, 0, stream>>>(Wq, Wqh, nW4);
  half_cvt<<<(nW4 + 255) / 256, 256, 0, stream>>>(Wk, Wkh, nW4);

  const int n4 = NB * CF * PPIX / 4;
  half_cvt<<<(n4 + 255) / 256, 256, 0, stream>>>(F, Vh, n4);

  proj_kernel<<<dim3(32, NB), 256, 0, stream>>>(X, Wqh, bq, Qh);
  proj_kernel<<<dim3(32, NB), 256, 0, stream>>>(F, Wkh, bk, Kh);

  flash_kernel<<<dim3(PPIX / 64, NB), 256, 0, stream>>>(Qh, Kh, Vh, out);
}
